// TransMIL_17746804867645
// MI455X (gfx1250) — matrix operand order repaired and checked by reading, not compiled, not run on hardware
//
#include <hip/hip_runtime.h>

// ---------------- problem constants ----------------
#define TOKN  22501      // tokens incl cls
#define TPAD  22528      // padded to multiple of 256 (and 64)
#define NPAD  27
#define NH    8
#define DHH   64
#define ML    256        // landmarks
#define LL    88         // tokens per landmark (TPAD/ML)
#define DC    512
#define NSPL  8          // flash split-K
#define HS    150

// gfx1250 async global->LDS staging (ASYNCcnt path). Set to 0 to fall back to
// the register-staged copy if the assembler rejects the mnemonic.
#define USE_ASYNC 1

// ---------------- bf16 helpers / WMMA ----------------
typedef __attribute__((ext_vector_type(16))) __bf16 v16bf;
typedef __attribute__((ext_vector_type(8)))  float  v8f;

struct __align__(16) U16B { unsigned int x[4]; };
union Frag16 { v16bf v; unsigned short u16[16]; U16B q[2]; };

__device__ __forceinline__ unsigned short f2bf(float f) {
    union { float f; unsigned int u; } v; v.f = f;
    return (unsigned short)((v.u + 0x7FFFu + ((v.u >> 16) & 1u)) >> 16);
}
__device__ __forceinline__ float bf2f(unsigned short s) {
    union { unsigned int u; float f; } v; v.u = ((unsigned int)s) << 16;
    return v.f;
}
__device__ __forceinline__ v8f wmma_bf16(const Frag16& a, const Frag16& b, v8f c) {
    return __builtin_amdgcn_wmma_f32_16x16x32_bf16(false, a.v, false, b.v, (short)0, c, false, false);
}

#if USE_ASYNC
// 16B async copy global -> LDS, tracked by ASYNCcnt (no VGPR round trip).
__device__ __forceinline__ void async_b128(const void* gptr, void* ldsptr) {
    unsigned lds = (unsigned)(unsigned long long)ldsptr;   // generic low 32b == LDS offset
    asm volatile("global_load_async_to_lds_b128 %0, %1, off"
                 :: "v"(lds), "v"(gptr) : "memory");
}
__device__ __forceinline__ void wait_async0() {
#if __has_builtin(__builtin_amdgcn_s_wait_asynccnt)
    __builtin_amdgcn_s_wait_asynccnt(0);
#else
    asm volatile("s_wait_asynccnt 0x0" ::: "memory");
#endif
}
#endif

// ---------------- generic bf16 WMMA GEMM ----------------
// C = alpha * A(MxK,row) @ B(KxN,row or N x K if bT) ; 64x64 tile, 4 waves, BK=32,
// double-buffered LDS with async global->LDS staging pipelined one chunk ahead.
// Preconditions: N%64==0, K%32==0, A rows readable up to ceil(M/64)*64.
__global__ __launch_bounds__(128) void gemm_bf16_kernel(
    const unsigned short* __restrict__ A, int lda, long long sA,
    const unsigned short* __restrict__ B, int ldb, long long sB, int bT,
    float* __restrict__ Cf, int ldc, long long sC,
    int M, int N, int K,
    const float* __restrict__ bias, const float* __restrict__ resid, int ldr,
    float alpha, int relu, int rshift, int Mout,
    int splitH, int headT, unsigned short* __restrict__ Chead)
{
    int tid = threadIdx.x, z = blockIdx.z;
    A += (long long)z * sA; B += (long long)z * sB;
    float* Cfz = Cf ? Cf + (long long)z * sC : (float*)0;
    int bm = blockIdx.y * 64, bn = blockIdx.x * 64;
    __shared__ __align__(16) unsigned short As[2][64 * 40];
    __shared__ __align__(16) unsigned short Bs[2][64 * 40];
    int wave = tid >> 5, lane = tid & 31;
    int wr = (wave & 1) * 32, wc = (wave >> 1) * 32;
    int rowl = lane & 15, koff = (lane >> 4) * 8, kb16 = (lane >> 4) * 16;

    auto stageA = [&](int kk, int bsel) {
        for (int i = 0; i < 2; i++) {
            int idx = tid * 2 + i, r = idx >> 2, c4 = idx & 3;
            const void* src = A + (long long)(bm + r) * lda + kk + c4 * 8;
#if USE_ASYNC
            async_b128(src, &As[bsel][r * 40 + c4 * 8]);
#else
            U16B t = *(const U16B*)src;
            *(U16B*)(&As[bsel][r * 40 + c4 * 8]) = t;
#endif
        }
    };
    auto stageB = [&](int kk, int bsel) {
        if (bT) {
            for (int i = 0; i < 2; i++) {
                int idx = tid * 2 + i, r = idx >> 2, c4 = idx & 3;
                const void* src = B + (long long)(bn + r) * ldb + kk + c4 * 8;
#if USE_ASYNC
                async_b128(src, &Bs[bsel][r * 40 + c4 * 8]);
#else
                U16B t = *(const U16B*)src;
                *(U16B*)(&Bs[bsel][r * 40 + c4 * 8]) = t;
#endif
            }
        } else {
            unsigned short tmp[16];
            for (int e = 0; e < 16; e++) {
                int idx = tid * 16 + e, k = idx >> 6, n = idx & 63;
                tmp[e] = B[(long long)(kk + k) * ldb + bn + n];
            }
            for (int e = 0; e < 16; e++) {
                int idx = tid * 16 + e, k = idx >> 6, n = idx & 63;
                Bs[bsel][n * 40 + k] = tmp[e];
            }
        }
    };

    v8f acc[2][2];
    for (int a0 = 0; a0 < 2; a0++) for (int b0 = 0; b0 < 2; b0++)
        for (int r = 0; r < 8; r++) acc[a0][b0][r] = 0.f;

    stageA(0, 0); stageB(0, 0);
#if USE_ASYNC
    wait_async0();
#endif
    __syncthreads();
    int buf = 0;
    for (int kk = 0; kk < K; kk += 32) {
        if (kk + 32 < K) {
            if (kk + 64 < K)
                __builtin_prefetch(A + (long long)(bm + (tid >> 1)) * lda + kk + 64, 0, 1);
            stageA(kk + 32, buf ^ 1);
            stageB(kk + 32, buf ^ 1);
        }
        Frag16 af[2], bf[2];
        for (int mt = 0; mt < 2; mt++) {
            int ar = wr + mt * 16 + rowl;
            af[mt].q[0] = *(const U16B*)(&As[buf][ar * 40 + koff]);
            af[mt].q[1] = *(const U16B*)(&As[buf][ar * 40 + 16 + koff]);
        }
        for (int nt = 0; nt < 2; nt++) {
            int bc = wc + nt * 16 + rowl;
            bf[nt].q[0] = *(const U16B*)(&Bs[buf][bc * 40 + koff]);
            bf[nt].q[1] = *(const U16B*)(&Bs[buf][bc * 40 + 16 + koff]);
        }
        for (int mt = 0; mt < 2; mt++)
            for (int nt = 0; nt < 2; nt++)
                acc[mt][nt] = wmma_bf16(af[mt], bf[nt], acc[mt][nt]);
#if USE_ASYNC
        wait_async0();
#endif
        __syncthreads();
        buf ^= 1;
    }
    int rbase = 8 * (lane >> 4), col0 = lane & 15;
    for (int mt = 0; mt < 2; mt++)
    for (int nt = 0; nt < 2; nt++)
    for (int r = 0; r < 8; r++) {
        int row = bm + wr + mt * 16 + rbase + r;
        int col = bn + wc + nt * 16 + col0;
        float v = acc[mt][nt][r] * alpha;
        if (splitH) {
            Chead[(long long)(col >> 6) * headT * 64 + (long long)row * 64 + (col & 63)] = f2bf(v);
        } else {
            int orow = row + rshift;
            if (orow < 0 || orow >= Mout) continue;
            if (bias)  v += bias[col];
            if (resid) v += resid[(long long)orow * ldr + col];
            if (relu)  v = v > 0.f ? v : 0.f;
            Cfz[(long long)orow * ldc + col] = v;
        }
    }
}

// ---------------- fused attn1: Merged[:,h*64:] = softmax(Q KL^T) @ W + conv ----------------
__global__ __launch_bounds__(128) void flash_attn1_kernel(
    const unsigned short* __restrict__ Q,   // [NH][T][64] (pre-scaled)
    const unsigned short* __restrict__ KL,  // [NH][256][64]
    const float* __restrict__ W,            // [NH][256][64]
    const float* __restrict__ conv,         // [NH][T][64]
    unsigned short* __restrict__ Merged,    // [T][512]
    int T)
{
    extern __shared__ __align__(16) char smem[];
    unsigned short* KLs = (unsigned short*)smem;            // [256][72]
    unsigned short* Wt  = KLs + 256 * 72;                   // [64][264] transposed
    float* Ssc  = (float*)(smem + 70656);                   // [64][257]
    float* rinv = (float*)(smem + 136448);                  // [64]
    int h = blockIdx.y, t0 = blockIdx.x * 64;
    int tid = threadIdx.x, wave = tid >> 5, lane = tid & 31;
    int rowl = lane & 15, koff = (lane >> 4) * 8, kb16 = (lane >> 4) * 16;
    int rbase = 8 * (lane >> 4), col0 = lane & 15;

    for (int i = 0; i < 16; i++) {                          // KL -> LDS (async b128)
        int idx = tid + i * 128, r = idx >> 3, c4 = idx & 7;
        const void* src = KL + ((long long)h * 256 + r) * 64 + c4 * 8;
#if USE_ASYNC
        async_b128(src, &KLs[r * 72 + c4 * 8]);
#else
        U16B t = *(const U16B*)src;
        *(U16B*)(&KLs[r * 72 + c4 * 8]) = t;
#endif
    }
    for (int e = 0; e < 128; e++) {                         // W -> LDS transposed bf16
        int idx = tid + e * 128, j = idx >> 6, d = idx & 63;
        Wt[d * 264 + j] = f2bf(W[(long long)h * 256 * 64 + idx]);
    }
#if USE_ASYNC
    wait_async0();
#endif
    __syncthreads();

    Frag16 qf[2];
    {
        const unsigned short* qp = Q + ((long long)h * T + t0 + wave * 16 + rowl) * 64;
        qf[0].q[0] = *(const U16B*)(qp + koff);
        qf[0].q[1] = *(const U16B*)(qp + 16 + koff);
        qf[1].q[0] = *(const U16B*)(qp + 32 + koff);
        qf[1].q[1] = *(const U16B*)(qp + 48 + koff);
    }
    for (int jt = 0; jt < 16; jt++) {
        v8f s; for (int r = 0; r < 8; r++) s[r] = 0.f;
        Frag16 b0, b1; int jr = jt * 16 + rowl;
        b0.q[0] = *(const U16B*)(&KLs[jr * 72 + koff]);
        b0.q[1] = *(const U16B*)(&KLs[jr * 72 + 16 + koff]);
        b1.q[0] = *(const U16B*)(&KLs[jr * 72 + 32 + koff]);
        b1.q[1] = *(const U16B*)(&KLs[jr * 72 + 48 + koff]);
        s = wmma_bf16(qf[0], b0, s);
        s = wmma_bf16(qf[1], b1, s);
        for (int r = 0; r < 8; r++)
            Ssc[(wave * 16 + rbase + r) * 257 + jt * 16 + col0] = s[r];
    }
    __syncthreads();
    if (tid < 64) {                                          // row softmax (unnormalized p; 1/sum kept)
        float mx = -1e30f;
        for (int j = 0; j < 256; j++) mx = fmaxf(mx, Ssc[tid * 257 + j]);
        float sum = 0.f;
        for (int j = 0; j < 256; j++) {
            float p = __expf(Ssc[tid * 257 + j] - mx);
            Ssc[tid * 257 + j] = p; sum += p;
        }
        rinv[tid] = 1.f / sum;
    }
    __syncthreads();

    v8f acc[4];
    for (int nt = 0; nt < 4; nt++) for (int r = 0; r < 8; r++) acc[nt][r] = 0.f;
    int ar = wave * 16 + rowl;
    for (int kc = 0; kc < 256; kc += 32) {
        Frag16 a;
        for (int i = 0; i < 8; i++) {
            a.u16[i]     = f2bf(Ssc[ar * 257 + kc + koff + i]);
            a.u16[8 + i] = f2bf(Ssc[ar * 257 + kc + 16 + koff + i]);
        }
        for (int nt = 0; nt < 4; nt++) {
            Frag16 b; int dc = nt * 16 + rowl;
            b.q[0] = *(const U16B*)(&Wt[dc * 264 + kc + koff]);
            b.q[1] = *(const U16B*)(&Wt[dc * 264 + kc + 16 + koff]);
            acc[nt] = wmma_bf16(a, b, acc[nt]);
        }
    }
    for (int nt = 0; nt < 4; nt++)
    for (int r = 0; r < 8; r++) {
        int row = t0 + wave * 16 + rbase + r, d = nt * 16 + col0;
        float v = acc[nt][r] * rinv[wave * 16 + rbase + r]
                + conv[((long long)h * T + row) * 64 + d];
        Merged[(long long)row * 512 + h * 64 + d] = f2bf(v);
    }
}

// ---------------- flash (attn3 @ v): online softmax over T, split-K partials ----------------
__global__ __launch_bounds__(128) void flash_kv_kernel(
    const unsigned short* __restrict__ QL,  // [NH][256][64]
    const unsigned short* __restrict__ Kb,  // [NH][T][64]
    const unsigned short* __restrict__ Vb,  // [NH][T][64]
    float* __restrict__ pAcc, float* __restrict__ pM, float* __restrict__ pL,
    int T, int nsplit)
{
    int split = blockIdx.x, mb = blockIdx.y, h = blockIdx.z;
    __shared__ __align__(16) unsigned short Ks[2][64 * 72];
    __shared__ __align__(16) unsigned short Vt[2][64 * 72];
    __shared__ float Ssc[64 * 65];
    __shared__ float mrow[64], lrow[64], rsc[64];
    int tid = threadIdx.x, wave = tid >> 5, lane = tid & 31;
    int rowl = lane & 15, koff = (lane >> 4) * 8, kb16 = (lane >> 4) * 16;
    int rbase = 8 * (lane >> 4), col0 = lane & 15;
    if (tid < 64) { mrow[tid] = -1e30f; lrow[tid] = 0.f; }

    auto stageKV = [&](int tb, int bsel) {
        for (int i = 0; i < 4; i++) {                       // K chunk -> LDS (async b128)
            int idx = tid + i * 128, r = idx >> 3, c4 = idx & 7;
            const void* src = Kb + ((long long)h * T + tb + r) * 64 + c4 * 8;
#if USE_ASYNC
            async_b128(src, &Ks[bsel][r * 72 + c4 * 8]);
#else
            U16B t = *(const U16B*)src;
            *(U16B*)(&Ks[bsel][r * 72 + c4 * 8]) = t;
#endif
        }
        unsigned short tv[32];                              // V chunk transposed (reg batch)
        for (int e = 0; e < 32; e++) {
            int idx = tid + e * 128, tl = idx >> 6, d = idx & 63;
            tv[e] = Vb[((long long)h * T + tb + tl) * 64 + d];
        }
        for (int e = 0; e < 32; e++) {
            int idx = tid + e * 128, tl = idx >> 6, d = idx & 63;
            Vt[bsel][d * 72 + tl] = tv[e];
        }
    };

    v8f acc[4];
    for (int nt = 0; nt < 4; nt++) for (int r = 0; r < 8; r++) acc[nt][r] = 0.f;
    Frag16 qf[2];
    {
        const unsigned short* qp = QL + ((long long)h * 256 + mb * 64 + wave * 16 + rowl) * 64;
        qf[0].q[0] = *(const U16B*)(qp + koff);
        qf[0].q[1] = *(const U16B*)(qp + 16 + koff);
        qf[1].q[0] = *(const U16B*)(qp + 32 + koff);
        qf[1].q[1] = *(const U16B*)(qp + 48 + koff);
    }
    int cl = T / nsplit, tstart = split * cl;
    stageKV(tstart, 0);
#if USE_ASYNC
    wait_async0();
#endif
    __syncthreads();
    int buf = 0;
    for (int tc = 0; tc < cl; tc += 64) {
        if (tc + 64 < cl) stageKV(tstart + tc + 64, buf ^ 1);
        for (int jt = 0; jt < 4; jt++) {
            v8f s; for (int r = 0; r < 8; r++) s[r] = 0.f;
            Frag16 b0, b1; int jr = jt * 16 + rowl;
            b0.q[0] = *(const U16B*)(&Ks[buf][jr * 72 + koff]);
            b0.q[1] = *(const U16B*)(&Ks[buf][jr * 72 + 16 + koff]);
            b1.q[0] = *(const U16B*)(&Ks[buf][jr * 72 + 32 + koff]);
            b1.q[1] = *(const U16B*)(&Ks[buf][jr * 72 + 48 + koff]);
            s = wmma_bf16(qf[0], b0, s);
            s = wmma_bf16(qf[1], b1, s);
            for (int r = 0; r < 8; r++)
                Ssc[(wave * 16 + rbase + r) * 65 + jt * 16 + col0] = s[r];
        }
        __syncthreads();
        if (tid < 64) {                                      // online update
            float m0 = mrow[tid], cm = m0;
            for (int j = 0; j < 64; j++) cm = fmaxf(cm, Ssc[tid * 65 + j]);
            float sc = __expf(m0 - cm);
            float l = lrow[tid] * sc;
            for (int j = 0; j < 64; j++) {
                float p = __expf(Ssc[tid * 65 + j] - cm);
                Ssc[tid * 65 + j] = p; l += p;
            }
            mrow[tid] = cm; lrow[tid] = l; rsc[tid] = sc;
        }
        __syncthreads();
        float scl[8];
        for (int r = 0; r < 8; r++) scl[r] = rsc[wave * 16 + rbase + r];
        for (int nt = 0; nt < 4; nt++)
            for (int r = 0; r < 8; r++) acc[nt][r] *= scl[r];
        int ar = wave * 16 + rowl;
        for (int kc = 0; kc < 64; kc += 32) {
            Frag16 a;
            for (int i = 0; i < 8; i++) {
                a.u16[i]     = f2bf(Ssc[ar * 65 + kc + koff + i]);
                a.u16[8 + i] = f2bf(Ssc[ar * 65 + kc + 16 + koff + i]);
            }
            for (int nt = 0; nt < 4; nt++) {
                Frag16 b; int dc = nt * 16 + rowl;
                b.q[0] = *(const U16B*)(&Vt[buf][dc * 72 + kc + koff]);
                b.q[1] = *(const U16B*)(&Vt[buf][dc * 72 + kc + 16 + koff]);
                acc[nt] = wmma_bf16(a, b, acc[nt]);
            }
        }
#if USE_ASYNC
        wait_async0();
#endif
        __syncthreads();
        buf ^= 1;
    }
    long long pb = ((long long)(h * 4 + mb) * nsplit + split);
    if (tid < 64) { pM[pb * 64 + tid] = mrow[tid]; pL[pb * 64 + tid] = lrow[tid]; }
    for (int nt = 0; nt < 4; nt++)
    for (int r = 0; r < 8; r++) {
        int row = wave * 16 + rbase + r, d = nt * 16 + col0;
        pAcc[(pb * 64 + row) * 64 + d] = acc[nt][r];
    }
}

__global__ void flash_kv_combine(const float* __restrict__ pAcc, const float* __restrict__ pM,
                                 const float* __restrict__ pL, float* __restrict__ B3, int nsplit)
{
    int b = blockIdx.x, h = b >> 8, m = b & 255, mb = m >> 6, r = m & 63, d = threadIdx.x;
    long long base = (long long)(h * 4 + mb) * nsplit;
    float mx = -1e30f;
    for (int s = 0; s < nsplit; s++) mx = fmaxf(mx, pM[(base + s) * 64 + r]);
    float L = 0.f, V = 0.f;
    for (int s = 0; s < nsplit; s++) {
        float w = __expf(pM[(base + s) * 64 + r] - mx);
        L += pL[(base + s) * 64 + r] * w;
        V += pAcc[((base + s) * 64 + r) * 64 + d] * w;
    }
    B3[((long long)h * 256 + m) * 64 + d] = V / L;
}

// ---------------- small fp32 batched GEMM (pinv chain, W=Z@B3) ----------------
__global__ __launch_bounds__(256) void gemm_f32_kernel(
    const float* __restrict__ A, const float* __restrict__ B, float* __restrict__ C,
    int M, int N, int K, float alpha, long long sA, long long sB, long long sC)
{
    int z = blockIdx.z;
    A += (long long)z * sA; B += (long long)z * sB; C += (long long)z * sC;
    int bm = blockIdx.y * 16, bn = blockIdx.x * 16;
    int tx = threadIdx.x & 15, ty = threadIdx.x >> 4;
    __shared__ float As[16][17], Bs[16][17];
    float acc = 0.f;
    for (int kk = 0; kk < K; kk += 16) {
        As[ty][tx] = A[(long long)(bm + ty) * K + kk + tx];
        Bs[ty][tx] = B[(long long)(kk + ty) * N + bn + tx];
        __syncthreads();
        for (int p = 0; p < 16; p++) acc += As[ty][p] * Bs[p][tx];
        __syncthreads();
    }
    C[(long long)(bm + ty) * N + bn + tx] = alpha * acc;
}

// ---------------- elementwise / reduction kernels ----------------
__global__ void cast_pad_kernel(const float* __restrict__ s, unsigned short* __restrict__ d,
                                long long ns, long long nd)
{
    long long i = (long long)blockIdx.x * 256 + threadIdx.x;
    if (i < nd) d[i] = (i < ns) ? f2bf(s[i]) : (unsigned short)0;
}

__global__ __launch_bounds__(256) void ln_kernel(const float* __restrict__ X,
    const float* __restrict__ w, const float* __restrict__ b,
    unsigned short* __restrict__ out, int padRows)
{
    int t = blockIdx.x, tid = threadIdx.x;
    unsigned short* orow = out + (long long)t * DC;
    if (t < padRows) { orow[tid] = 0; orow[tid + 256] = 0; return; }
    const float* x = X + (long long)(t - padRows) * DC;
    float a = x[tid], c = x[tid + 256];
    __shared__ float s1[256], s2[256];
    s1[tid] = a + c; s2[tid] = a * a + c * c; __syncthreads();
    for (int o = 128; o > 0; o >>= 1) {
        if (tid < o) { s1[tid] += s1[tid + o]; s2[tid] += s2[tid + o]; }
        __syncthreads();
    }
    float mu = s1[0] * (1.f / DC);
    float rs = rsqrtf(s2[0] * (1.f / DC) - mu * mu + 1e-5f);
    orow[tid]       = f2bf((a - mu) * rs * w[tid] + b[tid]);
    orow[tid + 256] = f2bf((c - mu) * rs * w[tid + 256] + b[tid + 256]);
}

__global__ void landmark_kernel(const unsigned short* __restrict__ Q,
    const unsigned short* __restrict__ K, unsigned short* __restrict__ QL,
    unsigned short* __restrict__ KL, int T)
{
    int b = blockIdx.x, h = b >> 8, m = b & 255, d = threadIdx.x;
    const unsigned short* qb = Q + ((long long)h * T + m * LL) * 64 + d;
    const unsigned short* kb = K + ((long long)h * T + m * LL) * 64 + d;
    float sq = 0.f, sk = 0.f;
    for (int j = 0; j < LL; j++) { sq += bf2f(qb[j * 64]); sk += bf2f(kb[j * 64]); }
    QL[((long long)h * 256 + m) * 64 + d] = f2bf(sq * (1.f / LL));
    KL[((long long)h * 256 + m) * 64 + d] = f2bf(sk * (1.f / LL));
}

__global__ __launch_bounds__(256) void softmax256_kernel(float* __restrict__ S)
{
    long long row = blockIdx.x; int tid = threadIdx.x;
    float v = S[row * 256 + tid];
    __shared__ float s[256];
    s[tid] = v; __syncthreads();
    for (int o = 128; o > 0; o >>= 1) { if (tid < o) s[tid] = fmaxf(s[tid], s[tid + o]); __syncthreads(); }
    float mx = s[0]; __syncthreads();
    float p = __expf(v - mx);
    s[tid] = p; __syncthreads();
    for (int o = 128; o > 0; o >>= 1) { if (tid < o) s[tid] += s[tid + o]; __syncthreads(); }
    S[row * 256 + tid] = p / s[0];
}

__global__ void abssum_kernel(const float* __restrict__ S, float* __restrict__ out, int transposed)
{
    int b = blockIdx.x, h = b >> 8, r = b & 255, tid = threadIdx.x;
    const float* base = S + (long long)h * 65536;
    float v = transposed ? fabsf(base[tid * 256 + r]) : fabsf(base[r * 256 + tid]);
    __shared__ float s[256];
    s[tid] = v; __syncthreads();
    for (int o = 128; o > 0; o >>= 1) { if (tid < o) s[tid] += s[tid + o]; __syncthreads(); }
    if (tid == 0) out[b] = s[0];
}

__global__ void reduce_max_kernel(const float* __restrict__ in, int n, float* __restrict__ out)
{
    int tid = threadIdx.x; float v = -1e30f;
    for (int i = tid; i < n; i += 256) v = fmaxf(v, in[i]);
    __shared__ float s[256];
    s[tid] = v; __syncthreads();
    for (int o = 128; o > 0; o >>= 1) { if (tid < o) s[tid] = fmaxf(s[tid], s[tid + o]); __syncthreads(); }
    if (tid == 0) out[0] = s[0];
}

__global__ void z0_kernel(const float* __restrict__ S, float* __restrict__ Z,
                          const float* __restrict__ rm, const float* __restrict__ cm, long long total)
{
    long long i = (long long)blockIdx.x * 256 + threadIdx.x;
    if (i >= total) return;
    float inv = 1.f / (rm[0] * cm[0]);
    long long hh = i >> 16; int r = (int)((i >> 8) & 255), c = (int)(i & 255);
    Z[i] = S[(hh << 16) + ((long long)c << 8) + r] * inv;
}

__global__ void dsub_kernel(float d, const float* __restrict__ A, float* __restrict__ C, long long total)
{
    long long i = (long long)blockIdx.x * 256 + threadIdx.x;
    if (i >= total) return;
    int r = (int)((i >> 8) & 255), c = (int)(i & 255);
    C[i] = (r == c ? d : 0.f) - A[i];
}

__global__ void conv_res_kernel(const unsigned short* __restrict__ V,
                                const float* __restrict__ resk, float* __restrict__ C, int T)
{
    long long i = (long long)blockIdx.x * 256 + threadIdx.x;
    long long tot = (long long)NH * T * 64;
    if (i >= tot) return;
    int h = (int)(i / ((long long)T * 64));
    long long rem = i - (long long)h * T * 64;
    int t = (int)(rem >> 6), d = (int)(rem & 63);
    float s = 0.f;
    for (int r = 0; r < 33; r++) {
        int tt = t + r - 16;
        if (tt >= 0 && tt < T) s += bf2f(V[((long long)h * T + tt) * 64 + d]) * resk[h * 33 + r];
    }
    C[i] = s;
}

__global__ void leff_kernel(const float* __restrict__ Xin, float* __restrict__ Xout,
    const float* __restrict__ k7, const float* __restrict__ b7,
    const float* __restrict__ k5, const float* __restrict__ b5,
    const float* __restrict__ k3, const float* __restrict__ b3)
{
    long long i = (long long)blockIdx.x * 256 + threadIdx.x;
    if (i >= (long long)HS * HS * DC) return;
    int p = (int)(i >> 9), c = (int)(i & 511);
    int hh = p / HS, ww = p % HS;
    float acc = Xin[(long long)(1 + p) * DC + c] + b7[c] + b5[c] + b3[c];
    for (int dy = 0; dy < 7; dy++) for (int dx = 0; dx < 7; dx++) {
        int yy = hh + dy - 3, xx = ww + dx - 3;
        if (yy >= 0 && yy < HS && xx >= 0 && xx < HS)
            acc += Xin[(long long)(1 + yy * HS + xx) * DC + c] * k7[c * 49 + dy * 7 + dx];
    }
    for (int dy = 0; dy < 5; dy++) for (int dx = 0; dx < 5; dx++) {
        int yy = hh + dy - 2, xx = ww + dx - 2;
        if (yy >= 0 && yy < HS && xx >= 0 && xx < HS)
            acc += Xin[(long long)(1 + yy * HS + xx) * DC + c] * k5[c * 25 + dy * 5 + dx];
    }
    for (int dy = 0; dy < 3; dy++) for (int dx = 0; dx < 3; dx++) {
        int yy = hh + dy - 1, xx = ww + dx - 1;
        if (yy >= 0 && yy < HS && xx >= 0 && xx < HS)
            acc += Xin[(long long)(1 + yy * HS + xx) * DC + c] * k3[c * 9 + dy * 3 + dx];
    }
    Xout[(long long)(1 + p) * DC + c] = acc;
}

__global__ void copy512_kernel(float* __restrict__ dst, const float* __restrict__ src)
{
    dst[threadIdx.x] = src[threadIdx.x];
}

// ---------------- cls-row path ----------------
__global__ __launch_bounds__(256) void row1_kernel(const unsigned short* __restrict__ Q0,
    const unsigned short* __restrict__ KL0, float* __restrict__ row1)
{
    int j = threadIdx.x;
    float s = 0.f;
    for (int d = 0; d < 64; d++) s += bf2f(Q0[NPAD * 64 + d]) * bf2f(KL0[j * 64 + d]);
    __shared__ float sm[256];
    sm[j] = s; __syncthreads();
    for (int o = 128; o > 0; o >>= 1) { if (j < o) sm[j] = fmaxf(sm[j], sm[j + o]); __syncthreads(); }
    float mx = sm[0]; __syncthreads();
    float p = __expf(s - mx);
    sm[j] = p; __syncthreads();
    for (int o = 128; o > 0; o >>= 1) { if (j < o) sm[j] += sm[j + o]; __syncthreads(); }
    row1[j] = p / sm[0];
}

__global__ void rowZ_kernel(const float* __restrict__ row1, const float* __restrict__ Z,
                            float* __restrict__ rowZ)
{
    int k = threadIdx.x; float s = 0.f;
    for (int j = 0; j < 256; j++) s += row1[j] * Z[j * 256 + k];
    rowZ[k] = s;
}

__global__ __launch_bounds__(256) void cls_stats_kernel(const unsigned short* __restrict__ QL0,
    const unsigned short* __restrict__ K0, float* __restrict__ mx, float* __restrict__ den, int T)
{
    int m = blockIdx.x, tid = threadIdx.x;
    __shared__ float qv[64];
    if (tid < 64) qv[tid] = bf2f(QL0[m * 64 + tid]);
    __syncthreads();
    float lm = -1e30f, ls = 0.f;
    for (int t = tid; t < T; t += 256) {
        float s = 0.f;
        for (int d = 0; d < 64; d++) s += qv[d] * bf2f(K0[(long long)t * 64 + d]);
        float nm = fmaxf(lm, s);
        ls = ls * __expf(lm - nm) + __expf(s - nm);
        lm = nm;
    }
    __shared__ float am[256], al[256];
    am[tid] = lm; al[tid] = ls; __syncthreads();
    for (int o = 128; o > 0; o >>= 1) {
        if (tid < o) {
            float m1 = am[tid], l1 = al[tid], m2 = am[tid + o], l2 = al[tid + o];
            float nm = fmaxf(m1, m2);
            al[tid] = l1 * __expf(m1 - nm) + l2 * __expf(m2 - nm);
            am[tid] = nm;
        }
        __syncthreads();
    }
    if (tid == 0) { mx[m] = am[0]; den[m] = al[0]; }
}

__global__ void coef_kernel(const float* __restrict__ rowZ, const float* __restrict__ den,
                            float* __restrict__ coef)
{
    coef[threadIdx.x] = rowZ[threadIdx.x] / den[threadIdx.x];
}

__global__ __launch_bounds__(256) void cls_out_kernel(const unsigned short* __restrict__ QL0,
    const unsigned short* __restrict__ K0, const float* __restrict__ coef,
    const float* __restrict__ mx, float* __restrict__ out, int T)
{
    int t = NPAD + 1 + blockIdx.x, m = threadIdx.x;
    __shared__ float kv[64];
    if (m < 64) kv[m] = bf2f(K0[(long long)t * 64 + m]);
    __syncthreads();
    float s = 0.f;
    for (int d = 0; d < 64; d++) s += bf2f(QL0[m * 64 + d]) * kv[d];
    float v = coef[m] * __expf(s - mx[m]);
    __shared__ float sm[256];
    sm[m] = v; __syncthreads();
    for (int o = 128; o > 0; o >>= 1) { if (m < o) sm[m] += sm[m + o]; __syncthreads(); }
    if (m == 0) out[blockIdx.x] = sm[0];
}

__global__ __launch_bounds__(256) void head_kernel(const float* __restrict__ h2,
    const float* __restrict__ nw, const float* __restrict__ nb,
    const float* __restrict__ fw, const float* __restrict__ fb, float* __restrict__ out)
{
    int tid = threadIdx.x;
    float a = h2[tid], c = h2[tid + 256];
    __shared__ float s1[256], s2[256];
    s1[tid] = a + c; s2[tid] = a * a + c * c; __syncthreads();
    for (int o = 128; o > 0; o >>= 1) { if (tid < o) { s1[tid] += s1[tid + o]; s2[tid] += s2[tid + o]; } __syncthreads(); }
    float mu = s1[0] * (1.f / DC);
    float rs = rsqrtf(s2[0] * (1.f / DC) - mu * mu + 1e-5f);
    float h0 = (a - mu) * rs * nw[tid] + nb[tid];
    float h1 = (c - mu) * rs * nw[tid + 256] + nb[tid + 256];
    __shared__ float r0[256], r1[256];
    r0[tid] = h0 * fw[tid * 2] + h1 * fw[(tid + 256) * 2];
    r1[tid] = h0 * fw[tid * 2 + 1] + h1 * fw[(tid + 256) * 2 + 1];
    __syncthreads();
    for (int o = 128; o > 0; o >>= 1) { if (tid < o) { r0[tid] += r0[tid + o]; r1[tid] += r1[tid + o]; } __syncthreads(); }
    if (tid == 0) {
        float l0 = r0[0] + fb[0], l1 = r1[0] + fb[1];
        float m = fmaxf(l0, l1);
        float e0 = __expf(l0 - m), e1 = __expf(l1 - m), inv = 1.f / (e0 + e1);
        out[0] = l0; out[1] = l1; out[2] = e0 * inv; out[3] = e1 * inv;
    }
}

// ==================== host orchestration ====================
struct WS {
    float *Xa, *Xb;
    unsigned short *LNbf, *hbf, *fc1wbf, *wqkv1, *wqkv2, *wout1, *wout2;
    unsigned short *Qb, *Kb, *Vb, *QLb, *KLb, *Merged;
    float *convb, *S2, *Zb, *Z2b, *XZb, *Tb, *Ub;
    float *rsum, *csum, *rc, *B3, *Wm, *pAcc, *pM, *pL;
    float *row1b, *rowZb, *coefb, *mxb, *denb;
};

static void launch_gemm(hipStream_t st, const unsigned short* A, int lda, long long sA,
                        const unsigned short* B, int ldb, long long sB, int bT,
                        float* Cf, int ldc, long long sC, int M, int N, int K,
                        const float* bias, const float* resid, int ldr,
                        float alpha, int relu, int rshift, int Mout,
                        int splitH, int headT, unsigned short* Chead, int batch)
{
    dim3 g(N / 64, (M + 63) / 64, batch);
    gemm_bf16_kernel<<<g, 128, 0, st>>>(A, lda, sA, B, ldb, sB, bT, Cf, ldc, sC,
                                        M, N, K, bias, resid, ldr, alpha, relu, rshift, Mout,
                                        splitH, headT, Chead);
}

static void launch_gemm_f32(hipStream_t st, const float* A, const float* B, float* C,
                            int M, int N, int K, float alpha, int batch)
{
    dim3 g(N / 16, M / 16, batch);
    gemm_f32_kernel<<<g, 256, 0, st>>>(A, B, C, M, N, K, alpha,
                                       (long long)M * K, (long long)K * N, (long long)M * N);
}

static float* run_pinv(hipStream_t st, const WS& w, int nh)
{
    abssum_kernel<<<nh * 256, 256, 0, st>>>(w.S2, w.rsum, 0);
    abssum_kernel<<<nh * 256, 256, 0, st>>>(w.S2, w.csum, 1);
    reduce_max_kernel<<<1, 256, 0, st>>>(w.rsum, nh * 256, w.rc);
    reduce_max_kernel<<<1, 256, 0, st>>>(w.csum, nh * 256, w.rc + 1);
    long long tot = (long long)nh * 65536;
    int nb = (int)((tot + 255) / 256);
    z0_kernel<<<nb, 256, 0, st>>>(w.S2, w.Zb, w.rc, w.rc + 1, tot);
    float* zc = w.Zb; float* za = w.Z2b;
    for (int it = 0; it < 6; it++) {
        launch_gemm_f32(st, w.S2, zc, w.XZb, 256, 256, 256, 1.f, nh);
        dsub_kernel<<<nb, 256, 0, st>>>(7.f, w.XZb, w.Tb, tot);
        launch_gemm_f32(st, w.XZb, w.Tb, w.Ub, 256, 256, 256, 1.f, nh);
        dsub_kernel<<<nb, 256, 0, st>>>(15.f, w.Ub, w.Tb, tot);
        launch_gemm_f32(st, w.XZb, w.Tb, w.Ub, 256, 256, 256, 1.f, nh);
        dsub_kernel<<<nb, 256, 0, st>>>(13.f, w.Ub, w.Tb, tot);
        launch_gemm_f32(st, zc, w.Tb, za, 256, 256, 256, 0.25f, nh);
        float* t = zc; zc = za; za = t;
    }
    return zc;
}

static void run_nystrom(hipStream_t st, const WS& w, const float* Xin, float* Xout,
                        const unsigned short* wqkvbf, const unsigned short* woutbf,
                        const float* bout, const float* resk,
                        const float* lnw, const float* lnb)
{
    ln_kernel<<<TPAD, 256, 0, st>>>(Xin, lnw, lnb, w.LNbf, NPAD);
    long long hseg = (long long)ML * 64;
    launch_gemm(st, w.LNbf, 512, 0, wqkvbf, 1536, 0, 0, 0, 0, 0, TPAD, 512, 512,
                0, 0, 0, 0.125f, 0, 0, 0, 1, TPAD, w.Qb, 1);
    launch_gemm(st, w.LNbf, 512, 0, wqkvbf + 512, 1536, 0, 0, 0, 0, 0, TPAD, 512, 512,
                0, 0, 0, 1.f, 0, 0, 0, 1, TPAD, w.Kb, 1);
    launch_gemm(st, w.LNbf, 512, 0, wqkvbf + 1024, 1536, 0, 0, 0, 0, 0, TPAD, 512, 512,
                0, 0, 0, 1.f, 0, 0, 0, 1, TPAD, w.Vb, 1);
    landmark_kernel<<<NH * ML, 64, 0, st>>>(w.Qb, w.Kb, w.QLb, w.KLb, TPAD);
    launch_gemm(st, w.QLb, 64, hseg, w.KLb, 64, hseg, 1, w.S2, 256, 65536,
                256, 256, 64, 0, 0, 0, 1.f, 0, 0, 256, 0, 0, 0, NH);
    softmax256_kernel<<<NH * ML, 256, 0, st>>>(w.S2);
    float* Zf = run_pinv(st, w, NH);
    flash_kv_kernel<<<dim3(NSPL, 4, NH), 128, 0, st>>>(w.QLb, w.Kb, w.Vb, w.pAcc, w.pM, w.pL, TPAD, NSPL);
    flash_kv_combine<<<NH * ML, 64, 0, st>>>(w.pAcc, w.pM, w.pL, w.B3, NSPL);
    launch_gemm_f32(st, Zf, w.B3, w.Wm, 256, 64, 256, 1.f, NH);
    long long ctot = (long long)NH * TPAD * 64;
    conv_res_kernel<<<(int)((ctot + 255) / 256), 256, 0, st>>>(w.Vb, resk, w.convb, TPAD);
    flash_attn1_kernel<<<dim3(TPAD / 64, NH), 128, 136704, st>>>(w.Qb, w.KLb, w.Wm, w.convb, w.Merged, TPAD);
    launch_gemm(st, w.Merged, 512, 0, woutbf, 512, 0, 0, Xout, 512, 0, TPAD, 512, 512,
                bout, Xin, 512, 1.f, 0, -NPAD, TOKN, 0, 0, 0, 1);
}

extern "C" void kernel_launch(void* const* d_in, const int* in_sizes, int n_in,
                              void* d_out, int out_size, void* d_ws, size_t ws_size,
                              hipStream_t stream)
{
    (void)in_sizes; (void)n_in; (void)out_size; (void)ws_size;
    const float* h       = (const float*)d_in[0];
    const float* fc1_w   = (const float*)d_in[1];
    const float* fc1_b   = (const float*)d_in[2];
    const float* cls_tok = (const float*)d_in[3];
    const float* l1_lnw  = (const float*)d_in[4];
    const float* l1_lnb  = (const float*)d_in[5];
    const float* l1_wqkv = (const float*)d_in[6];
    const float* l1_wout = (const float*)d_in[7];
    const float* l1_bout = (const float*)d_in[8];
    const float* l1_resk = (const float*)d_in[9];
    const float* l2_lnw  = (const float*)d_in[10];
    const float* l2_lnb  = (const float*)d_in[11];
    const float* l2_wqkv = (const float*)d_in[12];
    const float* l2_wout = (const float*)d_in[13];
    const float* l2_bout = (const float*)d_in[14];
    const float* l2_resk = (const float*)d_in[15];
    const float* k7 = (const float*)d_in[16]; const float* b7 = (const float*)d_in[17];
    const float* k5 = (const float*)d_in[18]; const float* b5 = (const float*)d_in[19];
    const float* k3 = (const float*)d_in[20]; const float* b3 = (const float*)d_in[21];
    const float* norm_w  = (const float*)d_in[22];
    const float* norm_b  = (const float*)d_in[23];
    const float* fc2_w   = (const float*)d_in[24];
    const float* fc2_b   = (const float*)d_in[25];
    float* out = (float*)d_out;

    char* base = (char*)d_ws; size_t off = 0;
    auto alloc = [&](size_t bytes) -> void* {
        off = (off + 255) & ~(size_t)255;
        void* p = base + off; off += bytes; return p;
    };
    WS w;
    w.Xa     = (float*)alloc((size_t)TOKN * DC * 4);
    w.Xb     = (float*)alloc((size_t)TOKN * DC * 4);
    w.LNbf   = (unsigned short*)alloc((size_t)TPAD * DC * 2);
    w.hbf    = (unsigned short*)alloc((size_t)TPAD * 1024 * 2);
    w.fc1wbf = (unsigned short*)alloc((size_t)1024 * 512 * 2);
    w.wqkv1  = (unsigned short*)alloc((size_t)512 * 1536 * 2);
    w.wqkv2  = (unsigned short*)alloc((size_t)512 * 1536 * 2);
    w.wout1  = (unsigned short*)alloc((size_t)512 * 512 * 2);
    w.wout2  = (unsigned short*)alloc((size_t)512 * 512 * 2);
    w.Qb     = (unsigned short*)alloc((size_t)TPAD * DC * 2);
    w.Kb     = (unsigned short*)alloc((size_t)TPAD * DC * 2);
    w.Vb     = (unsigned short*)alloc((size_t)TPAD * DC * 2);
    w.QLb    = (unsigned short*)alloc((size_t)NH * ML * 64 * 2);
    w.KLb    = (unsigned short*)alloc((size_t)NH * ML * 64 * 2);
    w.Merged = (unsigned short*)alloc((size_t)TPAD * DC * 2);
    w.convb  = (float*)alloc((size_t)NH * TPAD * 64 * 4);
    w.S2     = (float*)alloc((size_t)NH * 65536 * 4);
    w.Zb     = (float*)alloc((size_t)NH * 65536 * 4);
    w.Z2b    = (float*)alloc((size_t)NH * 65536 * 4);
    w.XZb    = (float*)alloc((size_t)NH * 65536 * 4);
    w.Tb     = (float*)alloc((size_t)NH * 65536 * 4);
    w.Ub     = (float*)alloc((size_t)NH * 65536 * 4);
    w.rsum   = (float*)alloc(2048 * 4);
    w.csum   = (float*)alloc(2048 * 4);
    w.rc     = (float*)alloc(2 * 4);
    w.B3     = (float*)alloc((size_t)NH * ML * 64 * 4);
    w.Wm     = (float*)alloc((size_t)NH * ML * 64 * 4);
    w.pAcc   = (float*)alloc((size_t)NH * 4 * NSPL * 64 * 64 * 4);
    w.pM     = (float*)alloc((size_t)NH * 4 * NSPL * 64 * 4);
    w.pL     = (float*)alloc((size_t)NH * 4 * NSPL * 64 * 4);
    w.row1b  = (float*)alloc(256 * 4);
    w.rowZb  = (float*)alloc(256 * 4);
    w.coefb  = (float*)alloc(256 * 4);
    w.mxb    = (float*)alloc(256 * 4);
    w.denb   = (float*)alloc(256 * 4);

    auto cast = [&](const float* s, unsigned short* d, long long ns, long long nd) {
        cast_pad_kernel<<<(int)((nd + 255) / 256), 256, 0, stream>>>(s, d, ns, nd);
    };
    // weight/input casts to bf16
    cast(h, w.hbf, (long long)22500 * 1024, (long long)TPAD * 1024);
    cast(fc1_w, w.fc1wbf, 1024 * 512, 1024 * 512);
    cast(l1_wqkv, w.wqkv1, 512 * 1536, 512 * 1536);
    cast(l2_wqkv, w.wqkv2, 512 * 1536, 512 * 1536);
    cast(l1_wout, w.wout1, 512 * 512, 512 * 512);
    cast(l2_wout, w.wout2, 512 * 512, 512 * 512);

    // fc1 + relu -> Xa rows 1..22500 ; cls token -> row 0
    launch_gemm(stream, w.hbf, 1024, 0, w.fc1wbf, 512, 0, 0, w.Xa, 512, 0,
                22500, 512, 1024, fc1_b, 0, 0, 1.f, 1, 1, TOKN, 0, 0, 0, 1);
    copy512_kernel<<<1, 512, 0, stream>>>(w.Xa, cls_tok);

    // layer 1
    run_nystrom(stream, w, w.Xa, w.Xb, w.wqkv1, w.wout1, l1_bout, l1_resk, l1_lnw, l1_lnb);
    // leff: Xb -> Xa (cls row copied)
    {
        long long tot = (long long)HS * HS * DC;
        leff_kernel<<<(int)((tot + 255) / 256), 256, 0, stream>>>(w.Xb, w.Xa, k7, b7, k5, b5, k3, b3);
        copy512_kernel<<<1, 512, 0, stream>>>(w.Xa, w.Xb);
    }
    // h2 = Xb, h3 = Xa
    run_nystrom(stream, w, w.Xa, w.Xb, w.wqkv2, w.wout2, l2_bout, l2_resk, l2_lnw, l2_lnb);
    run_nystrom(stream, w, w.Xb, w.Xa, w.wqkv2, w.wout2, l2_bout, l2_resk, l2_lnw, l2_lnb);

    // cls-row path on LN(h3) with l2 weights (head 0)
    ln_kernel<<<TPAD, 256, 0, stream>>>(w.Xa, l2_lnw, l2_lnb, w.LNbf, NPAD);
    launch_gemm(stream, w.LNbf, 512, 0, w.wqkv2, 1536, 0, 0, 0, 0, 0, TPAD, 512, 512,
                0, 0, 0, 0.125f, 0, 0, 0, 1, TPAD, w.Qb, 1);
    launch_gemm(stream, w.LNbf, 512, 0, w.wqkv2 + 512, 1536, 0, 0, 0, 0, 0, TPAD, 512, 512,
                0, 0, 0, 1.f, 0, 0, 0, 1, TPAD, w.Kb, 1);
    landmark_kernel<<<NH * ML, 64, 0, stream>>>(w.Qb, w.Kb, w.QLb, w.KLb, TPAD);
    launch_gemm(stream, w.QLb, 64, 0, w.KLb, 64, 0, 1, w.S2, 256, 0,
                256, 256, 64, 0, 0, 0, 1.f, 0, 0, 256, 0, 0, 0, 1);
    softmax256_kernel<<<ML, 256, 0, stream>>>(w.S2);
    float* Zc = run_pinv(stream, w, 1);
    row1_kernel<<<1, 256, 0, stream>>>(w.Qb, w.KLb, w.row1b);
    rowZ_kernel<<<1, 256, 0, stream>>>(w.row1b, Zc, w.rowZb);
    cls_stats_kernel<<<ML, 256, 0, stream>>>(w.QLb, w.Kb, w.mxb, w.denb, TPAD);
    coef_kernel<<<1, 256, 0, stream>>>(w.rowZb, w.denb, w.coefb);
    cls_out_kernel<<<22500, 256, 0, stream>>>(w.QLb, w.Kb, w.coefb, w.mxb, out + 4, TPAD);

    // logits / probs from layernorm(h2)[0]
    head_kernel<<<1, 256, 0, stream>>>(w.Xb, norm_w, norm_b, fc2_w, fc2_b, out);
}
